// PlannerIQGP_58643483460101
// MI455X (gfx1250) — matrix operand order repaired and checked by reading, not compiled, not run on hardware
//
#include <hip/hip_runtime.h>
#include <hip/hip_bf16.h>
#include <math.h>

// ---------------- types ----------------
typedef __attribute__((ext_vector_type(16))) __bf16 bf16x16;
typedef __attribute__((ext_vector_type(8)))  __bf16 bf16x8;
typedef __attribute__((ext_vector_type(8)))  float  floatx8;

static __device__ inline floatx8 wmma_bf16(bf16x16 a, bf16x16 b, floatx8 c) {
  return __builtin_amdgcn_wmma_f32_16x16x32_bf16(false, a, false, b, (short)0, c, false, false);
}

// 16-lane-group reductions (wave32: xor masks < 16 stay within the half)
static __device__ inline float rmax16(float v) {
  for (int off = 8; off; off >>= 1) v = fmaxf(v, __shfl_xor(v, off, 32));
  return v;
}
static __device__ inline float rsum16(float v) {
  for (int off = 8; off; off >>= 1) v += __shfl_xor(v, off, 32);
  return v;
}

// ---------------- problem constants ----------------
constexpr int Bc = 8, Sc = 1024, Hc = 1024, NHc = 8, Nc = 256, Kc = 8192, Ec = 2048, Uc = 512, HDc = 128;

// ---------------- elementwise kernels ----------------
__global__ void k_cvt_bf16(const float* __restrict__ in, __bf16* __restrict__ out, long n) {
  for (long i = (long)blockIdx.x * blockDim.x + threadIdx.x; i < n; i += (long)gridDim.x * blockDim.x)
    out[i] = (__bf16)in[i];
}

__global__ void k_transpose_cvt(const float* __restrict__ in, __bf16* __restrict__ out, int R, int C) {
  long i = (long)blockIdx.x * blockDim.x + threadIdx.x;
  if (i < (long)R * C) {
    int r = (int)(i / C), c = (int)(i % C);
    out[(long)c * R + r] = (__bf16)in[(long)r * C + c];
  }
}

__global__ void k_combine_half(const float* __restrict__ a, const float* __restrict__ b,
                               __bf16* __restrict__ out, long n) {
  for (long i = (long)blockIdx.x * blockDim.x + threadIdx.x; i < n; i += (long)gridDim.x * blockDim.x)
    out[i] = (__bf16)(0.5f * (a[i] + b[i]));
}

__global__ void k_zero(float* p, int n) {
  int i = blockIdx.x * blockDim.x + threadIdx.x;
  if (i < n) p[i] = 0.f;
}

__global__ void k_mask_neginf(float* __restrict__ lg, const int* __restrict__ mask, int NN, int EE, long n) {
  long i = (long)blockIdx.x * blockDim.x + threadIdx.x;
  if (i < n) {
    int e = (int)(i % EE);
    int b = (int)(i / ((long)NN * EE));
    if (!mask[(long)b * EE + e]) lg[i] = -__builtin_inff();
  }
}

__global__ void k_edge_epilogue(float* __restrict__ lg, const float* __restrict__ eb,
                                const float* __restrict__ ebias, int NN, long n) {
  long i = (long)blockIdx.x * blockDim.x + threadIdx.x;
  if (i < n) {
    long nm = i % ((long)NN * NN);
    lg[i] += eb[0] + ebias[nm];
  }
}

// ---------------- generic bf16 WMMA GEMM:  C = act(A@W^T + bias + accum) ----------------
// A: [M,K] bf16 row-major (lda, batch stride sA)   W: [N,K] bf16 row-major (ldw, batch stride sW)
// wave -> 16x64 output tile (A fragment reused across 4 WMMAs); block = 2x4 waves -> 32x256 tile
__global__ __launch_bounds__(256) void k_gemm(
    const __bf16* __restrict__ A, int lda, long sA,
    const __bf16* __restrict__ W, int ldw, long sW,
    const float* __restrict__ biasv,
    const float* __restrict__ accumIn,
    float* __restrict__ outF, __bf16* __restrict__ outB, int ldc, long sC,
    int Ksize, int act) {
  const int wv = threadIdx.x >> 5, lane = threadIdx.x & 31;
  const int half = lane >> 4, r = lane & 15;
  const int wrow = wv >> 2, wcol = wv & 3;
  const long b = blockIdx.z;
  const int row0 = blockIdx.y * 32 + wrow * 16;
  const int col0 = blockIdx.x * 256 + wcol * 64;

  const __bf16* ap = A + b * sA + (long)(row0 + r) * lda + (half ? 16 : 0);
  const __bf16* wp = W + b * sW + (long)(col0 + r) * ldw + (half ? 16 : 0);
  const long wstep = (long)16 * ldw;

  floatx8 acc[4];
#pragma unroll
  for (int s = 0; s < 4; s++)
#pragma unroll
    for (int j = 0; j < 8; j++) acc[s][j] = 0.f;

  for (int k = 0; k < Ksize; k += 32) {
    // pull the next fragment lines toward the WGP while WMMAs run (global_prefetch_b8)
    __builtin_prefetch(ap + k + 256, 0, 3);
    __builtin_prefetch(wp + k + 256, 0, 3);
    bf16x8 a0 = *(const bf16x8*)(ap + k);
    bf16x8 a1 = *(const bf16x8*)(ap + k + 8);
    bf16x16 af;
#pragma unroll
    for (int i = 0; i < 8; i++) { af[i] = a0[i]; af[i + 8] = a1[i]; }
#pragma unroll
    for (int s = 0; s < 4; s++) {
      bf16x16 wf = *(const bf16x16*)(wp + s * wstep + k);
      acc[s] = wmma_bf16(af, wf, acc[s]);
    }
  }

  const long cbase = b * sC;
#pragma unroll
  for (int s = 0; s < 4; s++) {
    const int col = col0 + s * 16 + r;
    const float bias = biasv ? biasv[col] : 0.f;
#pragma unroll
    for (int j = 0; j < 8; j++) {
      int row = row0 + half * 8 + j;
      long idx = cbase + (long)row * ldc + col;
      float v = acc[s][j] + bias;
      if (accumIn) v += accumIn[idx];
      if (act == 1) v = fmaxf(v, 0.f);
      else if (act == 2) v = tanhf(v);
      if (outF) outF[idx] = v;
      if (outB) outB[idx] = (__bf16)v;
    }
  }
}

// ---------------- flash attention: 8 waves/block, shared K+V LDS staging ----------------
// Q: [.,H] bf16 (per-batch stride sQ, may be 0), K/V: rows of [.,ldkv] bf16, O: bf16 [.,ldo]
// block = 256 threads = 8 waves; each wave owns one 16-query tile (128 queries per block).
// Per 32-key chunk: V staged transposed (Vt[d][key], B-frags for P@V are contiguous reads),
// K staged raw (Kl[key][d], B-frags for Q@K^T are contiguous reads). Two barriers per chunk;
// each wave reads only its OWN P tile from LDS, so no mid-chunk cross-wave barrier is needed.
__global__ __launch_bounds__(256) void k_attn(
    const __bf16* __restrict__ Qb, long sQ, int ldq,
    const __bf16* __restrict__ Kb, const __bf16* __restrict__ Vb, int ldkv, long sKV,
    __bf16* __restrict__ Ob, int ldo, long sO,
    const int* __restrict__ mask, int Skeys, float scale) {
  const int wv = threadIdx.x >> 5, lane = threadIdx.x & 31;
  const int half = lane >> 4, r = lane & 15;
  const int q0 = blockIdx.x * 128 + wv * 16, h = blockIdx.y, b = blockIdx.z;
  const int hoff = h * HDc;

  const __bf16* Q = Qb + (long)b * sQ;
  const __bf16* Kp = Kb + (long)b * sKV;
  const __bf16* Vp = Vb + (long)b * sKV;

  __shared__ __bf16 Vt[HDc * 48];     // transposed V chunk: [d][key], row stride 48
  __shared__ __bf16 Kl[32 * 144];     // raw K chunk: [key][d], row stride 144 (32B-aligned rows)
  __shared__ __bf16 Pl[8][16 * 32];   // per-wave probability tile [qrow][key]

  // preload Q fragments (A-matrix layout), contraction dim = HD = 4 steps of 32
  bf16x16 qf[4];
  {
    const __bf16* qp = Q + (long)(q0 + r) * ldq + hoff + (half ? 16 : 0);
#pragma unroll
    for (int t = 0; t < 4; t++) {
      bf16x8 lo = *(const bf16x8*)(qp + t * 32);
      bf16x8 hi = *(const bf16x8*)(qp + t * 32 + 8);
#pragma unroll
      for (int i = 0; i < 8; i++) { qf[t][i] = lo[i]; qf[t][i + 8] = hi[i]; }
    }
  }

  float m8[8], l8[8];
  floatx8 o[8];
#pragma unroll
  for (int j = 0; j < 8; j++) { m8[j] = -3.0e38f; l8[j] = 0.f; }
#pragma unroll
  for (int dt = 0; dt < 8; dt++)
#pragma unroll
    for (int j = 0; j < 8; j++) o[dt][j] = 0.f;

  const int tkey = threadIdx.x & 31, tdg = threadIdx.x >> 5;  // staging split: key row x 16-d group

  for (int s0 = 0; s0 < Skeys; s0 += 32) {
    // ---- cooperative staging: 256 threads cover 32 keys x 128 d for both K and V ----
    {
      const __bf16* vp = Vp + (long)(s0 + tkey) * ldkv + hoff + tdg * 16;
      const __bf16* kp = Kp + (long)(s0 + tkey) * ldkv + hoff + tdg * 16;
      __builtin_prefetch(vp + 32 * ldkv, 0, 3);   // next chunk (global_prefetch_b8)
      __builtin_prefetch(kp + 32 * ldkv, 0, 3);
      bf16x16 vvv = *(const bf16x16*)(vp);
#pragma unroll
      for (int i = 0; i < 16; i++) Vt[(tdg * 16 + i) * 48 + tkey] = vvv[i];  // transpose V
      *(bf16x16*)(&Kl[tkey * 144 + tdg * 16]) = *(const bf16x16*)(kp);       // raw K
    }
    __syncthreads();  // Kl + Vt fully staged

    // ---- scores for two 16-key subtiles (B-frags from LDS Kl, contiguous) ----
    float sA[2][8];
#pragma unroll
    for (int sub = 0; sub < 2; sub++) {
      floatx8 acc = {0.f, 0.f, 0.f, 0.f, 0.f, 0.f, 0.f, 0.f};
      const __bf16* kp = &Kl[(sub * 16 + r) * 144 + (half ? 16 : 0)];
#pragma unroll
      for (int t = 0; t < 4; t++) {
        bf16x16 kf = *(const bf16x16*)(kp + t * 32);
        acc = wmma_bf16(qf[t], kf, acc);
      }
      float km = 0.f;
      if (mask) {
        int key = s0 + sub * 16 + r;  // C layout: col = lane&15
        km = mask[(long)b * Skeys + key] ? 0.f : -1.0e30f;
      }
#pragma unroll
      for (int j = 0; j < 8; j++) sA[sub][j] = acc[j] * scale + km;
    }

    // ---- online softmax per q-row ----
#pragma unroll
    for (int j = 0; j < 8; j++) {
      float mx = rmax16(fmaxf(sA[0][j], sA[1][j]));
      float nm = fmaxf(m8[j], mx);
      float corr = __expf(m8[j] - nm);
      m8[j] = nm;
      float p0 = __expf(sA[0][j] - nm);
      float p1 = __expf(sA[1][j] - nm);
      l8[j] = l8[j] * corr + rsum16(p0 + p1);
#pragma unroll
      for (int dt = 0; dt < 8; dt++) o[dt][j] *= corr;
      int row = half * 8 + j;
      Pl[wv][row * 32 + r] = (__bf16)p0;
      Pl[wv][row * 32 + 16 + r] = (__bf16)p1;
    }

    // ---- P(16x32) @ V(32x128): wave-local P (contiguous k order), shared Vt ----
    bf16x16 pf;
    {
      const __bf16* pp = &Pl[wv][r * 32 + (half ? 16 : 0)];
#pragma unroll
      for (int i = 0; i < 8; i++) { pf[i] = pp[i]; pf[i + 8] = pp[i + 8]; }
    }
#pragma unroll
    for (int dt = 0; dt < 8; dt++) {
      bf16x16 vf = *(const bf16x16*)(&Vt[(dt * 16 + r) * 48 + half * 16]);
      o[dt] = wmma_bf16(pf, vf, o[dt]);
    }
    __syncthreads();  // protect Kl/Vt before next iteration's staging
  }

  // normalize + store (C layout: col = lane&15, row = half*8+j)
#pragma unroll
  for (int j = 0; j < 8; j++) {
    float inv = l8[j] > 0.f ? 1.f / l8[j] : 0.f;
    int row = q0 + half * 8 + j;
#pragma unroll
    for (int dt = 0; dt < 8; dt++)
      Ob[(long)b * sO + (long)row * ldo + hoff + dt * 16 + r] = (__bf16)(o[dt][j] * inv);
  }
}

// ---------------- LayerNorm (block per row) ----------------
__global__ __launch_bounds__(256) void k_ln(const float* __restrict__ in, const float* __restrict__ g,
                                            const float* __restrict__ bt, float* __restrict__ outF,
                                            __bf16* __restrict__ outB, int cols) {
  __shared__ float red[256];
  const long row = blockIdx.x;
  const int tid = threadIdx.x;
  float v[4];
  float s = 0.f;
#pragma unroll
  for (int i = 0; i < 4; i++) { v[i] = in[row * cols + tid + i * 256]; s += v[i]; }
  red[tid] = s; __syncthreads();
  for (int st = 128; st; st >>= 1) { if (tid < st) red[tid] += red[tid + st]; __syncthreads(); }
  float mu = red[0] / cols; __syncthreads();
  float s2 = 0.f;
#pragma unroll
  for (int i = 0; i < 4; i++) { float d = v[i] - mu; s2 += d * d; }
  red[tid] = s2; __syncthreads();
  for (int st = 128; st; st >>= 1) { if (tid < st) red[tid] += red[tid + st]; __syncthreads(); }
  float rstd = rsqrtf(red[0] / cols + 1e-5f);
#pragma unroll
  for (int i = 0; i < 4; i++) {
    int c = tid + i * 256;
    float ov = (v[i] - mu) * rstd * g[c] + bt[c];
    outF[row * cols + c] = ov;
    outB[row * cols + c] = (__bf16)ov;
  }
}

// ---------------- codebook row norms ----------------
__global__ __launch_bounds__(256) void k_rownorm2(const float* __restrict__ in, float* __restrict__ out, int cols) {
  __shared__ float red[256];
  const long row = blockIdx.x;
  const int tid = threadIdx.x;
  float s = 0.f;
  for (int c = tid; c < cols; c += 256) { float x = in[row * cols + c]; s += x * x; }
  red[tid] = s; __syncthreads();
  for (int st = 128; st; st >>= 1) { if (tid < st) red[tid] += red[tid + st]; __syncthreads(); }
  if (tid == 0) out[row] = red[0];
}

// ---------------- fused VQ argmin over a 512-code column group (WMMA dot tiles) ----------------
// wave: 16 node rows; scans its group in 64-code strips with 4 accumulators (A fragment reused)
__global__ __launch_bounds__(32) void k_vq(const __bf16* __restrict__ X, const __bf16* __restrict__ CB,
                                           const float* __restrict__ cbn, float* __restrict__ pmin,
                                           int* __restrict__ pidx, int Kdim, int colsPerGroup) {
  const int lane = threadIdx.x;
  const int half = lane >> 4, r = lane & 15;
  const int row0 = blockIdx.y * 16;
  const int cg = blockIdx.x * colsPerGroup;
  const __bf16* ap = X + (long)(row0 + r) * Kdim + (half ? 16 : 0);
  const long wstep = (long)16 * Kdim;

  float bm[8]; int bi[8];
#pragma unroll
  for (int j = 0; j < 8; j++) { bm[j] = 3.0e38f; bi[j] = 0; }

  for (int c0 = cg; c0 < cg + colsPerGroup; c0 += 64) {
    const __bf16* wp = CB + (long)(c0 + r) * Kdim + (half ? 16 : 0);
    floatx8 acc[4];
#pragma unroll
    for (int s = 0; s < 4; s++)
#pragma unroll
      for (int j = 0; j < 8; j++) acc[s][j] = 0.f;
    for (int k = 0; k < Kdim; k += 32) {
      __builtin_prefetch(wp + k + 256, 0, 3);
      bf16x8 a0 = *(const bf16x8*)(ap + k);
      bf16x8 a1 = *(const bf16x8*)(ap + k + 8);
      bf16x16 af;
#pragma unroll
      for (int i = 0; i < 8; i++) { af[i] = a0[i]; af[i + 8] = a1[i]; }
#pragma unroll
      for (int s = 0; s < 4; s++) {
        bf16x16 wf = *(const bf16x16*)(wp + s * wstep + k);
        acc[s] = wmma_bf16(af, wf, acc[s]);
      }
    }
#pragma unroll
    for (int s = 0; s < 4; s++) {
      int col = c0 + s * 16 + r;
      float cn = cbn[col];
#pragma unroll
      for (int j = 0; j < 8; j++) {
        float v = cn - 2.f * acc[s][j];
        if (v < bm[j]) { bm[j] = v; bi[j] = col; }
      }
    }
  }
  // reduce across the 16 lanes of each half (argmin with lowest-index tiebreak)
  for (int off = 8; off; off >>= 1) {
#pragma unroll
    for (int j = 0; j < 8; j++) {
      float ov = __shfl_xor(bm[j], off, 32);
      int oi = __shfl_xor(bi[j], off, 32);
      if (ov < bm[j] || (ov == bm[j] && oi < bi[j])) { bm[j] = ov; bi[j] = oi; }
    }
  }
  if (r == 0) {
#pragma unroll
    for (int j = 0; j < 8; j++) {
      int row = row0 + half * 8 + j;
      pmin[(long)row * gridDim.x + blockIdx.x] = bm[j];
      pidx[(long)row * gridDim.x + blockIdx.x] = bi[j];
    }
  }
}

__global__ void k_vq_reduce(const float* __restrict__ pmin, const int* __restrict__ pidx, int groups,
                            int rows, float* __restrict__ codesF, int* __restrict__ codesI) {
  int row = blockIdx.x * blockDim.x + threadIdx.x;
  if (row < rows) {
    float bm = 3.0e38f; int bi = 0;
    for (int g = 0; g < groups; g++) {
      float v = pmin[(long)row * groups + g];
      int idx = pidx[(long)row * groups + g];
      if (v < bm) { bm = v; bi = idx; }
    }
    codesI[row] = bi;
    codesF[row] = (float)bi;
  }
}

__global__ void k_vq_gather(const int* __restrict__ codes, const float* __restrict__ cb,
                            float* __restrict__ qout, __bf16* __restrict__ qbf, long n, int Hd) {
  for (long i = (long)blockIdx.x * blockDim.x + threadIdx.x; i < n; i += (long)gridDim.x * blockDim.x) {
    long row = i / Hd; int hh = (int)(i % Hd);
    float v = cb[(long)codes[row] * Hd + hh];
    qout[i] = v;
    qbf[i] = (__bf16)v;
  }
}

__global__ __launch_bounds__(256) void k_vq_loss(const float* __restrict__ nodes, const float* __restrict__ quant,
                                                 float* __restrict__ loss, long n, float invn) {
  __shared__ float red[256];
  float s = 0.f;
  for (long i = (long)blockIdx.x * blockDim.x + threadIdx.x; i < n; i += (long)gridDim.x * blockDim.x) {
    float d = nodes[i] - quant[i];
    s += d * d;
  }
  red[threadIdx.x] = s; __syncthreads();
  for (int st = 128; st; st >>= 1) { if (threadIdx.x < st) red[threadIdx.x] += red[threadIdx.x + st]; __syncthreads(); }
  if (threadIdx.x == 0) atomicAdd(loss, red[0] * invn);
}

// ---------------- length controller: ll = t . w2 + b2 ; mask = ll > 0 ----------------
__global__ __launch_bounds__(256) void k_len_mask(const float* __restrict__ t, const float* __restrict__ w2,
                                                  const float* __restrict__ b2, float* __restrict__ outM, int cols) {
  __shared__ float red[256];
  const long row = blockIdx.x;
  const int tid = threadIdx.x;
  float s = 0.f;
  for (int c = tid; c < cols; c += 256) s += t[row * cols + c] * w2[c];
  red[tid] = s; __syncthreads();
  for (int st = 128; st; st >>= 1) { if (tid < st) red[tid] += red[tid + st]; __syncthreads(); }
  if (tid == 0) outM[row] = (red[0] + b2[0]) > 0.f ? 1.f : 0.f;
}

// =====================================================================================
extern "C" void kernel_launch(void* const* d_in, const int* in_sizes, int n_in,
                              void* d_out, int out_size, void* d_ws, size_t ws_size,
                              hipStream_t stream) {
  (void)in_sizes; (void)n_in; (void)out_size; (void)ws_size;

  // ---- inputs (setup_inputs() order) ----
  const float* h_en      = (const float*)d_in[0];
  const float* h_zh      = (const float*)d_in[1];
  const float* ent_emb   = (const float*)d_in[2];
  const int*   ent_mask  = (const int*)d_in[3];
  const float* unit_emb  = (const float*)d_in[4];
  const int*   unit_mask = (const int*)d_in[5];
  const int*   tok_mask  = (const int*)d_in[6];
  const float* en_in_w = (const float*)d_in[7],  *en_in_b = (const float*)d_in[8];
  const float* en_out_w= (const float*)d_in[9],  *en_out_b= (const float*)d_in[10];
  const float* zh_in_w = (const float*)d_in[11], *zh_in_b = (const float*)d_in[12];
  const float* zh_out_w= (const float*)d_in[13], *zh_out_b= (const float*)d_in[14];
  const float* nd_in_w = (const float*)d_in[15], *nd_in_b = (const float*)d_in[16];
  const float* nd_out_w= (const float*)d_in[17], *nd_out_b= (const float*)d_in[18];
  const float* fuse_w1 = (const float*)d_in[19], *fuse_b1 = (const float*)d_in[20];
  const float* fuse_w2 = (const float*)d_in[21], *fuse_b2 = (const float*)d_in[22];
  const float* node_q  = (const float*)d_in[23];
  const float* norm_g  = (const float*)d_in[24], *norm_b  = (const float*)d_in[25];
  const float* codebook= (const float*)d_in[26];
  const float* ent_ptr_w = (const float*)d_in[27];
  const float* unit_ptr_w= (const float*)d_in[28];
  const float* edge_w  = (const float*)d_in[29];
  const float* edge_b  = (const float*)d_in[30];
  const float* edge_bias=(const float*)d_in[31];
  const float* len_w1  = (const float*)d_in[32], *len_b1 = (const float*)d_in[33];
  const float* len_w2  = (const float*)d_in[34], *len_b2 = (const float*)d_in[35];

  // ---- output layout (flat f32, tuple order) ----
  float* out = (float*)d_out;
  const long O_QUANT = 0;
  const long O_CODES = O_QUANT + (long)Bc * Nc * Hc;          // 2,097,152
  const long O_QID   = O_CODES + (long)Bc * Nc;               // +2048
  const long O_UNIT  = O_QID + (long)Bc * Nc * Ec;            // +4,194,304
  const long O_EDGE  = O_UNIT + (long)Bc * Nc * Uc;           // +1,048,576
  const long O_MASK  = O_EDGE + (long)Bc * Nc * Nc;           // +524,288
  const long O_LOSS  = O_MASK + (long)Bc * Nc;                // +2048

  // ---- workspace bump allocator ----
  size_t off = 0;
  auto wsa = [&](size_t bytes) -> void* {
    void* p = (char*)d_ws + off;
    off += (bytes + 255) & ~(size_t)255;
    return p;
  };
  const long SH = (long)Sc * Hc;          // 1,048,576
  const long BSH = (long)Bc * SH;         // 8,388,608
  const long BNH = (long)Bc * Nc * Hc;    // 2,097,152

  __bf16* X_en  = (__bf16*)wsa(BSH * 2);
  __bf16* X_zh  = (__bf16*)wsa(BSH * 2);
  __bf16* ENT   = (__bf16*)wsa((long)Bc * Ec * Hc * 2);
  __bf16* UNIT  = (__bf16*)wsa((long)Bc * Uc * Hc * 2);
  __bf16* Win_en = (__bf16*)wsa((long)3 * Hc * Hc * 2);
  __bf16* Win_zh = (__bf16*)wsa((long)3 * Hc * Hc * 2);
  __bf16* Win_nd = (__bf16*)wsa((long)3 * Hc * Hc * 2);
  __bf16* Wout_en= (__bf16*)wsa((long)Hc * Hc * 2);
  __bf16* Wout_zh= (__bf16*)wsa((long)Hc * Hc * 2);
  __bf16* Wout_nd= (__bf16*)wsa((long)Hc * Hc * 2);
  __bf16* Wf1   = (__bf16*)wsa((long)Hc * 2 * Hc * 2);
  __bf16* Wf2   = (__bf16*)wsa((long)Hc * Hc * 2);
  __bf16* CBbf  = (__bf16*)wsa((long)Kc * Hc * 2);
  __bf16* Wentp = (__bf16*)wsa((long)Hc * Hc * 2);
  __bf16* Wunitp= (__bf16*)wsa((long)Hc * Hc * 2);
  __bf16* WedgeT= (__bf16*)wsa((long)Hc * Hc * 2);
  __bf16* Wlen1 = (__bf16*)wsa((long)Hc * Hc * 2);
  __bf16* NQ    = (__bf16*)wsa((long)Nc * Hc * 2);
  __bf16* Q_en  = (__bf16*)wsa(BSH * 2);
  __bf16* KV_en = (__bf16*)wsa(BSH * 2 * 2);
  __bf16* Q_zh  = (__bf16*)wsa(BSH * 2);
  __bf16* KV_zh = (__bf16*)wsa(BSH * 2 * 2);
  __bf16* AO_en = (__bf16*)wsa(BSH * 2);
  __bf16* AO_zh = (__bf16*)wsa(BSH * 2);
  __bf16* ATT_en= (__bf16*)wsa(BSH * 2);
  __bf16* ATT_zh= (__bf16*)wsa(BSH * 2);
  __bf16* HID   = (__bf16*)wsa(BSH * 2);
  float*  GHID  = (float*)wsa(BSH * 4);
  float*  G_en  = (float*)wsa(BSH * 4);
  float*  G_zh  = (float*)wsa(BSH * 4);
  __bf16* FUSED = (__bf16*)wsa(BSH * 2);
  __bf16* QN    = (__bf16*)wsa((long)Nc * Hc * 2);
  __bf16* KV_nd = (__bf16*)wsa(BSH * 2 * 2);
  __bf16* AO_nd = (__bf16*)wsa(BNH * 2);
  float*  NPRE  = (float*)wsa(BNH * 4);
  float*  NODES = (float*)wsa(BNH * 4);
  __bf16* NODESb= (__bf16*)wsa(BNH * 2);
  float*  CBN   = (float*)wsa((long)Kc * 4);
  float*  PMIN  = (float*)wsa((long)Bc * Nc * 16 * 4);
  int*    PIDX  = (int*)wsa((long)Bc * Nc * 16 * 4);
  int*    CODES = (int*)wsa((long)Bc * Nc * 4);
  __bf16* QBF   = (__bf16*)wsa(BNH * 2);
  __bf16* TMPP  = (__bf16*)wsa(BNH * 2);
  __bf16* EQ    = (__bf16*)wsa(BNH * 2);
  float*  TANH  = (float*)wsa(BNH * 4);

  auto cvt = [&](const float* src, __bf16* dst, long n) {
    int blocks = (int)((n + 255) / 256);
    k_cvt_bf16<<<blocks, 256, 0, stream>>>(src, dst, n);
  };
  auto gemm = [&](const __bf16* A, int lda, long sA, const __bf16* W, int ldw, long sW,
                  const float* bias, const float* accumIn, float* outF, __bf16* outB,
                  int ldc, long sC, int M, int N, int Kd, int batch, int act) {
    dim3 g(N / 256, M / 32, batch);
    k_gemm<<<g, 256, 0, stream>>>(A, lda, sA, W, ldw, sW, bias, accumIn, outF, outB, ldc, sC, Kd, act);
  };

  // ---- phase 0: convert everything heavy to bf16 ----
  cvt(h_en, X_en, BSH);
  cvt(h_zh, X_zh, BSH);
  cvt(ent_emb, ENT, (long)Bc * Ec * Hc);
  cvt(unit_emb, UNIT, (long)Bc * Uc * Hc);
  cvt(en_in_w, Win_en, (long)3 * Hc * Hc);
  cvt(zh_in_w, Win_zh, (long)3 * Hc * Hc);
  cvt(nd_in_w, Win_nd, (long)3 * Hc * Hc);
  cvt(en_out_w, Wout_en, (long)Hc * Hc);
  cvt(zh_out_w, Wout_zh, (long)Hc * Hc);
  cvt(nd_out_w, Wout_nd, (long)Hc * Hc);
  cvt(fuse_w1, Wf1, (long)Hc * 2 * Hc);
  cvt(fuse_w2, Wf2, (long)Hc * Hc);
  cvt(codebook, CBbf, (long)Kc * Hc);
  cvt(ent_ptr_w, Wentp, (long)Hc * Hc);
  cvt(unit_ptr_w, Wunitp, (long)Hc * Hc);
  cvt(len_w1, Wlen1, (long)Hc * Hc);
  cvt(node_q, NQ, (long)Nc * Hc);
  k_transpose_cvt<<<(int)(((long)Hc * Hc + 255) / 256), 256, 0, stream>>>(edge_w, WedgeT, Hc, Hc);

  const float att_scale = 0.08838834764831845f;  // 1/sqrt(128)

  // ---- phase 1: QKV projections for cross attention ----
  gemm(X_en, Hc, 0, Win_en, Hc, 0, en_in_b, nullptr, nullptr, Q_en, Hc, 0, Bc * Sc, Hc, Hc, 1, 0);
  gemm(X_zh, Hc, 0, Win_en + (long)Hc * Hc, Hc, 0, en_in_b + Hc, nullptr, nullptr, KV_en, 2 * Hc, 0, Bc * Sc, 2 * Hc, Hc, 1, 0);
  gemm(X_zh, Hc, 0, Win_zh, Hc, 0, zh_in_b, nullptr, nullptr, Q_zh, Hc, 0, Bc * Sc, Hc, Hc, 1, 0);
  gemm(X_en, Hc, 0, Win_zh + (long)Hc * Hc, Hc, 0, zh_in_b + Hc, nullptr, nullptr, KV_zh, 2 * Hc, 0, Bc * Sc, 2 * Hc, Hc, 1, 0);

  // ---- phase 2: cross attention ----
  {
    dim3 g(Sc / 128, NHc, Bc);
    k_attn<<<g, 256, 0, stream>>>(Q_en, SH, Hc, KV_en, KV_en + Hc, 2 * Hc, (long)Sc * 2 * Hc,
                                  AO_en, Hc, SH, nullptr, Sc, att_scale);
    k_attn<<<g, 256, 0, stream>>>(Q_zh, SH, Hc, KV_zh, KV_zh + Hc, 2 * Hc, (long)Sc * 2 * Hc,
                                  AO_zh, Hc, SH, nullptr, Sc, att_scale);
  }

  // ---- phase 3: attention output projections ----
  gemm(AO_en, Hc, 0, Wout_en, Hc, 0, en_out_b, nullptr, nullptr, ATT_en, Hc, 0, Bc * Sc, Hc, Hc, 1, 0);
  gemm(AO_zh, Hc, 0, Wout_zh, Hc, 0, zh_out_b, nullptr, nullptr, ATT_zh, Hc, 0, Bc * Sc, Hc, Hc, 1, 0);

  // ---- phase 4: shared gate MLP, fused = 0.5*(gate(en)+gate(zh)) ----
  // gate(x,a): hidden = relu(x@W1x^T + a@W1a^T + b1); g = hidden@W2^T + b2
  gemm(X_en, Hc, 0, Wf1, 2 * Hc, 0, fuse_b1, nullptr, GHID, nullptr, Hc, 0, Bc * Sc, Hc, Hc, 1, 0);
  gemm(ATT_en, Hc, 0, Wf1 + Hc, 2 * Hc, 0, nullptr, GHID, nullptr, HID, Hc, 0, Bc * Sc, Hc, Hc, 1, 1);
  gemm(HID, Hc, 0, Wf2, Hc, 0, fuse_b2, nullptr, G_en, nullptr, Hc, 0, Bc * Sc, Hc, Hc, 1, 0);
  gemm(X_zh, Hc, 0, Wf1, 2 * Hc, 0, fuse_b1, nullptr, GHID, nullptr, Hc, 0, Bc * Sc, Hc, Hc, 1, 0);
  gemm(ATT_zh, Hc, 0, Wf1 + Hc, 2 * Hc, 0, nullptr, GHID, nullptr, HID, Hc, 0, Bc * Sc, Hc, Hc, 1, 1);
  gemm(HID, Hc, 0, Wf2, Hc, 0, fuse_b2, nullptr, G_zh, nullptr, Hc, 0, Bc * Sc, Hc, Hc, 1, 0);
  k_combine_half<<<(int)((BSH + 255) / 256), 256, 0, stream>>>(G_en, G_zh, FUSED, BSH);

  // ---- phase 5: node attention + out proj + LN ----
  gemm(NQ, Hc, 0, Win_nd, Hc, 0, nd_in_b, nullptr, nullptr, QN, Hc, 0, Nc, Hc, Hc, 1, 0);
  gemm(FUSED, Hc, 0, Win_nd + (long)Hc * Hc, Hc, 0, nd_in_b + Hc, nullptr, nullptr, KV_nd, 2 * Hc, 0, Bc * Sc, 2 * Hc, Hc, 1, 0);
  {
    dim3 g(Nc / 128, NHc, Bc);
    k_attn<<<g, 256, 0, stream>>>(QN, 0, Hc, KV_nd, KV_nd + Hc, 2 * Hc, (long)Sc * 2 * Hc,
                                  AO_nd, Hc, (long)Nc * Hc, tok_mask, Sc, att_scale);
  }
  gemm(AO_nd, Hc, 0, Wout_nd, Hc, 0, nd_out_b, nullptr, NPRE, nullptr, Hc, 0, Bc * Nc, Hc, Hc, 1, 0);
  k_ln<<<Bc * Nc, 256, 0, stream>>>(NPRE, norm_g, norm_b, NODES, NODESb, Hc);

  // ---- phase 6: VQ ----
  k_rownorm2<<<Kc, 256, 0, stream>>>(codebook, CBN, Hc);
  {
    dim3 g(16, (Bc * Nc) / 16, 1);  // 16 column groups of 512 codes
    k_vq<<<g, 32, 0, stream>>>(NODESb, CBbf, CBN, PMIN, PIDX, Hc, Kc / 16);
  }
  k_vq_reduce<<<(Bc * Nc + 255) / 256, 256, 0, stream>>>(PMIN, PIDX, 16, Bc * Nc, out + O_CODES, CODES);
  k_vq_gather<<<(int)((BNH + 255) / 256), 256, 0, stream>>>(CODES, codebook, out + O_QUANT, QBF, BNH, Hc);
  k_zero<<<1, 1, 0, stream>>>(out + O_LOSS, 1);
  k_vq_loss<<<1024, 256, 0, stream>>>(NODES, out + O_QUANT, out + O_LOSS, BNH, 1.0f / (float)BNH);

  // ---- phase 7: pointer logits ----
  gemm(QBF, Hc, 0, Wentp, Hc, 0, nullptr, nullptr, nullptr, TMPP, Hc, 0, Bc * Nc, Hc, Hc, 1, 0);
  gemm(TMPP, Hc, (long)Nc * Hc, ENT, Hc, (long)Ec * Hc, nullptr, nullptr, out + O_QID, nullptr,
       Ec, (long)Nc * Ec, Nc, Ec, Hc, Bc, 0);
  k_mask_neginf<<<(int)(((long)Bc * Nc * Ec + 255) / 256), 256, 0, stream>>>(out + O_QID, ent_mask, Nc, Ec, (long)Bc * Nc * Ec);
  gemm(QBF, Hc, 0, Wunitp, Hc, 0, nullptr, nullptr, nullptr, TMPP, Hc, 0, Bc * Nc, Hc, Hc, 1, 0);
  gemm(TMPP, Hc, (long)Nc * Hc, UNIT, Hc, (long)Uc * Hc, nullptr, nullptr, out + O_UNIT, nullptr,
       Uc, (long)Nc * Uc, Nc, Uc, Hc, Bc, 0);
  k_mask_neginf<<<(int)(((long)Bc * Nc * Uc + 255) / 256), 256, 0, stream>>>(out + O_UNIT, unit_mask, Nc, Uc, (long)Bc * Nc * Uc);

  // ---- phase 8: edge logits ----
  gemm(QBF, Hc, 0, WedgeT, Hc, 0, nullptr, nullptr, nullptr, EQ, Hc, 0, Bc * Nc, Hc, Hc, 1, 0);
  gemm(EQ, Hc, (long)Nc * Hc, QBF, Hc, (long)Nc * Hc, nullptr, nullptr, out + O_EDGE, nullptr,
       Nc, (long)Nc * Nc, Nc, Nc, Hc, Bc, 0);
  k_edge_epilogue<<<(int)(((long)Bc * Nc * Nc + 255) / 256), 256, 0, stream>>>(out + O_EDGE, edge_b, edge_bias, Nc, (long)Bc * Nc * Nc);

  // ---- phase 9: length controller -> node mask ----
  gemm(NODESb, Hc, 0, Wlen1, Hc, 0, len_b1, nullptr, TANH, nullptr, Hc, 0, Bc * Nc, Hc, Hc, 1, 2);
  k_len_mask<<<Bc * Nc, 256, 0, stream>>>(TANH, len_w2, len_b2, out + O_MASK, Hc);
}
